// LabelGRU_65747359367842
// MI455X (gfx1250) — matrix operand order repaired and checked by reading, not compiled, not run on hardware
//
#include <hip/hip_runtime.h>
#include <hip/hip_bf16.h>

// L=16 labels, D=256, H=128, B=32, T=2048, 3H=384
#define L_ 16
#define D_ 256
#define H_ 128
#define B_ 32
#define T_ 2048
#define G3 384

typedef __attribute__((ext_vector_type(16))) __bf16 v16bf;
typedef __attribute__((ext_vector_type(8)))  float  v8f;
typedef __attribute__((ext_vector_type(4)))  float  f32x4;
typedef __attribute__((ext_vector_type(4)))  unsigned int u32x4;
typedef __attribute__((ext_vector_type(4)))  int i32x4;

union BFrag { u32x4 u[2]; v16bf v; };

#if defined(__gfx1250__) && __has_builtin(__builtin_amdgcn_global_load_async_to_lds_b128)
#define HAS_ASYNC_LDS 1
#else
#define HAS_ASYNC_LDS 0
#endif

// 16-byte global -> LDS copy. On gfx1250 use the ASYNCcnt-tracked async path
// (GLOBAL_LOAD_ASYNC_TO_LDS_B128); otherwise a synchronous fallback.
// Builtin prototype (from compiler diagnostic): v4i AS1* src, v4i AS3* lds,
// imm offset, imm cpol.
__device__ __forceinline__ void async_cp16(const void* g, void* l) {
#if HAS_ASYNC_LDS
    __builtin_amdgcn_global_load_async_to_lds_b128(
        (__attribute__((address_space(1))) i32x4*)(unsigned long long)(uintptr_t)g,
        (__attribute__((address_space(3))) i32x4*)(unsigned int)(uintptr_t)l,
        0, 0);
#else
    *(u32x4*)l = *(const u32x4*)g;
#endif
}

__device__ __forceinline__ void async_wait0() {
#if HAS_ASYNC_LDS
#if __has_builtin(__builtin_amdgcn_s_wait_asynccnt)
    __builtin_amdgcn_s_wait_asynccnt(0);
#else
    asm volatile("s_wait_asynccnt 0" ::: "memory");
#endif
#endif
}

__device__ __forceinline__ v8f splat8(float s) {
    v8f v;
#pragma unroll
    for (int e = 0; e < 8; ++e) v[e] = s;
    return v;
}

__device__ __forceinline__ float sigm(float x) { return 1.0f / (1.0f + __expf(-x)); }
__device__ __forceinline__ float tanh_f(float x) { return 2.0f / (1.0f + __expf(-2.0f * x)) - 1.0f; }

// Build bf16 A-fragment (16x32, MxK) from four aligned float4s:
// per ISA layout a lane holds K runs {klo..klo+7} and {klo+16..klo+23}.
__device__ __forceinline__ v16bf mk_afrag(f32x4 a, f32x4 b, f32x4 c, f32x4 d) {
    v16bf v;
#pragma unroll
    for (int e = 0; e < 4; ++e) {
        v[e]      = (__bf16)a[e];
        v[4 + e]  = (__bf16)b[e];
        v[8 + e]  = (__bf16)c[e];
        v[12 + e] = (__bf16)d[e];
    }
    return v;
}

// Load bf16 B-fragment (32x16, KxN): lane holds K runs {klo..klo+7} and {klo+16..klo+23}.
__device__ __forceinline__ v16bf ldB(const __bf16* p) {
    BFrag f;
    f.u[0] = *(const u32x4*)p;
    f.u[1] = *(const u32x4*)(p + 16);
    return f.v;
}

__device__ __forceinline__ v8f wmma_bf16(v16bf a, v16bf b, v8f c) {
    return __builtin_amdgcn_wmma_f32_16x16x32_bf16(false, a, false, b, (short)0, c,
                                                   false, false);
}

// ---------------------------------------------------------------------------
// Phase 0: one-time weight conversion f32 -> bf16, label-mapping baked in.
// wih_bf[l,dir] = (bf16)Wih[label[l],dir], whh_bf likewise.
// ---------------------------------------------------------------------------
__global__ void __launch_bounds__(256) wcvt_kernel(
    const float* __restrict__ Wih, const float* __restrict__ Whh,
    const long long* __restrict__ label,
    __bf16* __restrict__ wih_bf, __bf16* __restrict__ whh_bf)
{
    const size_t NIH = (size_t)L_ * 2 * G3 * D_;
    const size_t NHH = (size_t)L_ * 2 * G3 * H_;
    size_t i = (size_t)blockIdx.x * 256 + threadIdx.x;
    if (i < NIH) {
        const size_t per = (size_t)2 * G3 * D_;
        int l = (int)(i / per);
        size_t r = i - (size_t)l * per;
        int lab = (int)label[l];
        wih_bf[i] = (__bf16)Wih[(size_t)lab * per + r];
    }
    if (i < NHH) {
        const size_t per = (size_t)2 * G3 * H_;
        int l = (int)(i / per);
        size_t r = i - (size_t)l * per;
        int lab = (int)label[l];
        whh_bf[i] = (__bf16)Whh[(size_t)lab * per + r];
    }
}

// ---------------------------------------------------------------------------
// Phase 1: Gi[l, tloc, b, g] = x[b, t, :] @ Wih[l,dir]^T + bih   (bf16 out)
// Full-chip GEMM, parallel over (l, t). Wih K-slices double-buffered in LDS
// via async-to-LDS copies. Block: 8 waves = 2 t x 2 batch-halves x 2 N-halves.
// ---------------------------------------------------------------------------
__global__ void __launch_bounds__(256) gi_gemm_kernel(
    const float* __restrict__ x,          // [B,T,D]
    const __bf16* __restrict__ wih_bf,    // [L,2,384,256] label-mapped
    const float* __restrict__ bih,        // [L,2,384]
    const long long* __restrict__ label,  // [L]
    __bf16* __restrict__ gi,              // [L,TC,32,384]
    int dir, int t0, int TC)
{
    __shared__ __bf16 sW[2][G3 * 40];  // 2 x (384 x 32) K-slices, rows padded to 40

    const int tpairs = TC >> 1;
    const int bid = blockIdx.x;
    const int l  = bid / tpairs;
    const int tp = bid - l * tpairs;
    const int lab = (int)label[l];

    const __bf16* gW = wih_bf + ((size_t)(l * 2 + dir)) * G3 * D_;
    const float*  gB = bih + (size_t)(lab * 2 + dir) * G3;

    const int tid  = threadIdx.x;
    const int lane = tid & 31;
    const int w    = tid >> 5;
    const int nloc = lane & 15;
    const int hi   = (lane >> 4) & 1;
    const int tw   = t0 + tp * 2 + (w & 1);
    const int hb   = (w >> 1) & 1;
    const int nh   = (w >> 2) & 1;

    const int brow = hb * 16 + nloc;  // A-matrix row (x row)
    const float* xrow = x + ((size_t)brow * T_ + tw) * D_ + hi * 8;

    // stage Wih[:, kb*32 .. kb*32+31] (bf16, raw bytes) -> LDS buf (async)
    auto stage = [&](int kb, int buf) {
        // 384 rows x 64B = 1536 chunks of 16B, 6 per thread, uniform trip count
        for (int c = tid; c < 1536; c += 256) {
            int g = c >> 2, q = c & 3;
            async_cp16(gW + (size_t)g * D_ + kb * 32 + q * 8,
                       &sW[buf][g * 40 + q * 8]);
        }
    };

    v8f acc[12];
#pragma unroll
    for (int i = 0; i < 12; ++i) acc[i] = splat8(0.0f);

    stage(0, 0);
    async_wait0();
    __syncthreads();

    for (int kb = 0; kb < 8; ++kb) {
        if (kb < 7) stage(kb + 1, (kb + 1) & 1);

        const __bf16* sWc = sW[kb & 1];
        const f32x4* px = (const f32x4*)(xrow + kb * 32);
        v16bf a = mk_afrag(px[0], px[1], px[4], px[5]);
        const int koff = hi * 8;

        // software-pipelined B-fragment loads feeding the WMMA chain
        v16bf bnx = ldB(&sWc[(nh * 192 + nloc) * 40 + koff]);
#pragma unroll
        for (int ct = 0; ct < 12; ++ct) {
            v16bf bb = bnx;
            if (ct < 11)
                bnx = ldB(&sWc[(nh * 192 + (ct + 1) * 16 + nloc) * 40 + koff]);
            acc[ct] = wmma_bf16(a, bb, acc[ct]);
        }
        async_wait0();   // next slice resident before the barrier releases it
        __syncthreads();
    }

    // epilogue: + bih, convert to bf16, store [l][tloc][b][g]
    const int tloc = tw - t0;
    __bf16* go = gi + ((size_t)l * TC + tloc) * 32 * G3;
#pragma unroll
    for (int ct = 0; ct < 12; ++ct) {
        int col = nh * 192 + ct * 16 + nloc;
        float bv = gB[col];
#pragma unroll
        for (int v = 0; v < 8; ++v) {
            int b = hb * 16 + hi * 8 + v;  // C/D layout: M = v (+8 for upper lanes)
            go[(size_t)b * G3 + col] = (__bf16)(acc[ct][v] + bv);
        }
    }
}

// ---------------------------------------------------------------------------
// Phase 2: sequential GRU scan over a chunk of T.
// 32 workgroups = 16 labels x 2 batch-halves (independent recurrent chains).
// Whh resident in LDS (bf16, padded rows); h state in LDS f32; per-step Gi
// tile double-buffered in LDS via async-to-LDS (prefetch next t during compute).
// Wave w owns h columns [16w,16w+16): gate tiles at N-offsets {16w,128+16w,256+16w}.
// ---------------------------------------------------------------------------
__global__ void __launch_bounds__(256) gru_scan_kernel(
    const __bf16* __restrict__ gi,        // [L,TC,32,384]
    const __bf16* __restrict__ whh_bf,    // [L,2,384,128] label-mapped
    const float* __restrict__ bhh,        // [L,2,384]
    const long long* __restrict__ label,  // [L]
    const int* __restrict__ amask,        // [B,T]
    float* __restrict__ hstate,           // [32][16][128] per-wg persistent h
    float* __restrict__ out,              // [B,L,256]
    int dir, int t0, int TC, int first, int last)
{
    __shared__ __bf16 sWhh[G3 * 136];      // rows padded 128 -> 136 (102 KB)
    __shared__ float  sH[16 * 132];        // rows padded 128 -> 132
    __shared__ __bf16 sGi[2][16 * G3];     // 2 x 12 KB per-step gate tiles
    __shared__ int    sLen[16];

    const int bid = blockIdx.x;
    const int l  = bid >> 1;
    const int hb = bid & 1;
    const int lab = (int)label[l];

    const __bf16* gW = whh_bf + ((size_t)(l * 2 + dir)) * G3 * H_;
    const float*  gb = bhh + (size_t)(lab * 2 + dir) * G3;

    const int tid  = threadIdx.x;
    const int lane = tid & 31;
    const int w    = tid >> 5;
    const int nloc = lane & 15;
    const int hi   = (lane >> 4) & 1;
    const int b0   = hb * 16;

    const __bf16* giL = gi + (size_t)l * TC * 32 * G3;

    // async stage of a step's 16x384 bf16 Gi tile (contiguous 12 KB)
    auto stageGi = [&](int tloc, int buf) {
        const __bf16* src = giL + (size_t)tloc * 32 * G3 + (size_t)b0 * G3;
        for (int c = tid; c < 768; c += 256)           // 3 x 16B per thread
            async_cp16(src + c * 8, &sGi[buf][c * 8]);
    };

    // stage Whh -> LDS (bf16 raw bytes, async), rows padded
    for (int c = tid; c < G3 * 16; c += 256) {         // 6144 chunks of 16B
        int g = c >> 4, q = c & 15;
        async_cp16(gW + (size_t)g * H_ + q * 8, &sWhh[g * 136 + q * 8]);
    }
    const int tfirst = dir ? (TC - 1) : 0;
    stageGi(tfirst, 0);

    // h init: zero on first chunk of a sweep, else restore from workspace
    float* hs = hstate + (size_t)bid * 16 * 128;
    for (int idx = tid; idx < 16 * 132; idx += 256) {
        int m = idx / 132, j = idx - m * 132;
        float v = 0.0f;
        if (!first && j < 128) v = hs[m * 128 + j];
        sH[idx] = v;
    }
    if (tid < 16) sLen[tid] = 0;
    __syncthreads();
    {   // lengths from prefix attention mask
        int r = tid & 15, part = tid >> 4;
        const int* mr = amask + (size_t)(b0 + r) * T_ + part * 128;
        int s = 0;
        for (int i = 0; i < 128; ++i) s += mr[i];
        atomicAdd(&sLen[r], s);
    }
    async_wait0();
    __syncthreads();

    const int j0 = w * 16;
    const float bhr = gb[j0 + nloc];
    const float bhz = gb[128 + j0 + nloc];
    const float bhn = gb[256 + j0 + nloc];
    int lenv[8];
#pragma unroll
    for (int v = 0; v < 8; ++v) lenv[v] = sLen[hi * 8 + v];

    for (int s = 0; s < TC; ++s) {
        const int tloc = dir ? (TC - 1 - s) : s;
        const int t = t0 + tloc;
        const int buf = s & 1;

        // kick async prefetch of next step's tile into the other buffer
        const int tn = dir ? (tloc - 1) : (tloc + 1);
        if (tn >= 0 && tn < TC) stageGi(tn, buf ^ 1);

        // accumulators seeded from LDS Gi tile (bih baked in) + bhh terms;
        // acc3: 0 = r-gate, 1 = z-gate, 2 = h_n path (kept separate for r*(.))
        v8f acc3[3], accin;
        const __bf16* gtile = sGi[buf];
#pragma unroll
        for (int v = 0; v < 8; ++v) {
            const int base = (hi * 8 + v) * G3;
            acc3[0][v] = (float)gtile[base + j0 + nloc] + bhr;
            acc3[1][v] = (float)gtile[base + 128 + j0 + nloc] + bhz;
            accin[v]   = (float)gtile[base + 256 + j0 + nloc];
            acc3[2][v] = bhn;
        }

        // gh = h @ Whh^T: A-fragments hoisted, B-fragments software-pipelined
        const float* hrow = &sH[nloc * 132 + hi * 8];
        v16bf afr[4];
#pragma unroll
        for (int kb = 0; kb < 4; ++kb) {
            const f32x4* ph = (const f32x4*)(hrow + kb * 32);
            afr[kb] = mk_afrag(ph[0], ph[1], ph[4], ph[5]);
        }
        const __bf16* baddr[12];
#pragma unroll
        for (int kb = 0; kb < 4; ++kb)
#pragma unroll
            for (int g = 0; g < 3; ++g)
                baddr[kb * 3 + g] =
                    &sWhh[(g * 128 + j0 + nloc) * 136 + kb * 32 + hi * 8];
        v16bf bnx = ldB(baddr[0]);
#pragma unroll
        for (int q = 0; q < 12; ++q) {
            v16bf bb = bnx;
            if (q < 11) bnx = ldB(baddr[q + 1]);
            acc3[q % 3] = wmma_bf16(afr[q / 3], bb, acc3[q % 3]);
        }

        float hold[8];
#pragma unroll
        for (int v = 0; v < 8; ++v)
            hold[v] = sH[(hi * 8 + v) * 132 + j0 + nloc];
        __syncthreads();  // all reads of h(t) complete

#pragma unroll
        for (int v = 0; v < 8; ++v) {
            float r = sigm(acc3[0][v]);
            float z = sigm(acc3[1][v]);
            float n = tanh_f(accin[v] + r * acc3[2][v]);
            float hn = (1.0f - z) * n + z * hold[v];
            float hv = (t < lenv[v]) ? hn : hold[v];
            sH[(hi * 8 + v) * 132 + j0 + nloc] = hv;
        }
        async_wait0();    // next Gi tile resident before releasing the barrier
        __syncthreads();  // h(t+1) + sGi[buf^1] visible to all waves
    }

    // persist h; on last chunk also emit output: out[b,l, bwd|fwd half]
    for (int idx = tid; idx < 16 * 128; idx += 256) {
        int m = idx >> 7, j = idx & 127;
        float hv = sH[m * 132 + j];
        hs[idx] = hv;
        if (last)
            out[((size_t)(b0 + m) * L_ + l) * 256 + (dir ? 0 : 128) + j] = hv;
    }
}

// ---------------------------------------------------------------------------
extern "C" void kernel_launch(void* const* d_in, const int* in_sizes, int n_in,
                              void* d_out, int out_size, void* d_ws, size_t ws_size,
                              hipStream_t stream)
{
    const float*     x     = (const float*)d_in[0];
    const int*       amask = (const int*)d_in[1];
    const long long* label = (const long long*)d_in[2];
    const float*     Wih   = (const float*)d_in[3];
    const float*     Whh   = (const float*)d_in[4];
    const float*     bih   = (const float*)d_in[5];
    const float*     bhh   = (const float*)d_in[6];
    float* out = (float*)d_out;

    // workspace layout: [hstate 256KB][wih_bf 6MB][whh_bf 3MB][gi chunk]
    const size_t hbytes   = (size_t)32 * 16 * 128 * sizeof(float);
    const size_t nih      = (size_t)L_ * 2 * G3 * D_;
    const size_t nhh      = (size_t)L_ * 2 * G3 * H_;
    float*  hstate = (float*)d_ws;
    __bf16* wih_bf = (__bf16*)((char*)d_ws + hbytes);
    __bf16* whh_bf = (__bf16*)((char*)wih_bf + nih * sizeof(__bf16));
    char*   gi_base = (char*)whh_bf + nhh * sizeof(__bf16);
    const size_t fixed = hbytes + (nih + nhh) * sizeof(__bf16);

    // Chunk T so the Gi staging buffer fits in workspace.
    int TC = 2048;
    while (TC > 2) {
        size_t need = fixed + (size_t)L_ * TC * 32 * G3 * sizeof(__bf16);
        if (need <= ws_size) break;
        TC >>= 1;
    }
    __bf16* gi = (__bf16*)gi_base;
    const int nch = 2048 / TC;

    wcvt_kernel<<<dim3((unsigned)(nih / 256)), 256, 0, stream>>>(
        Wih, Whh, label, wih_bf, whh_bf);

    for (int dir = 0; dir < 2; ++dir) {
        for (int ci = 0; ci < nch; ++ci) {
            const int c  = dir ? (nch - 1 - ci) : ci;  // bwd walks chunks in reverse
            const int t0 = c * TC;
            gi_gemm_kernel<<<dim3(L_ * (TC / 2)), 256, 0, stream>>>(
                x, wih_bf, bih, label, gi, dir, t0, TC);
            gru_scan_kernel<<<dim3(32), 256, 0, stream>>>(
                gi, whh_bf, bhh, label, amask, hstate, out, dir, t0, TC,
                ci == 0, ci == nch - 1);
        }
    }
}
